// Encoder_86947317940934
// MI455X (gfx1250) — matrix operand order repaired and checked by reading, not compiled, not run on hardware
//
#include <hip/hip_runtime.h>
#include <hip/hip_bf16.h>

// ---------------- CDNA5 vector types ----------------
typedef __bf16 v16bf __attribute__((ext_vector_type(16)));
typedef __bf16 v8bf  __attribute__((ext_vector_type(8)));
typedef float  v8f   __attribute__((ext_vector_type(8)));

#define N_NODES  50000
#define N_EDGES  800000
#define F_IN     128
#define DIM      256
#define N_LAYERS 3
#define N_GRAPHS 512
#define OUT_DIM  (N_LAYERS * DIM)   // 768

// ---------------- small utility kernels ----------------
__global__ void zero_f32(float* __restrict__ p, long long n) {
    long long i = (long long)blockIdx.x * blockDim.x + threadIdx.x;
    if (i < n) p[i] = 0.0f;
}

__global__ void copy_f4(const float* __restrict__ in, float* __restrict__ out, int n4) {
    int i = blockIdx.x * blockDim.x + threadIdx.x;
    if (i < n4) ((float4*)out)[i] = ((const float4*)in)[i];
}

// w: [K x 256] row-major fp32  ->  wt: [256 x K] row-major bf16 (i.e. N-major)
__global__ void wt_transpose_bf16(const float* __restrict__ w, __bf16* __restrict__ wt, int K) {
    int idx = blockIdx.x * blockDim.x + threadIdx.x;       // coalesced read of w
    if (idx >= K * DIM) return;
    int k = idx / DIM;
    int n = idx % DIM;
    wt[(size_t)n * K + k] = (__bf16)w[idx];
}

// ---------------- edge scatter: agg[dst] += h[src] ----------------
// one thread = one (edge, 4-feature chunk); lanes cover consecutive chunks
__global__ void edge_scatter(const float* __restrict__ h,
                             const int* __restrict__ src,
                             const int* __restrict__ dst,
                             float* __restrict__ agg,
                             int fin4, int total) {
    int g = blockIdx.x * blockDim.x + threadIdx.x;
    if (g >= total) return;
    int e = g / fin4;
    int c = g - e * fin4;
    int s = src[e];
    int d = dst[e];
    float4 v = ((const float4*)(h + (size_t)s * fin4 * 4))[c];
    float* o = agg + (size_t)d * fin4 * 4 + c * 4;
    atomicAdd(o + 0, v.x);
    atomicAdd(o + 1, v.y);
    atomicAdd(o + 2, v.z);
    atomicAdd(o + 3, v.w);
}

// ---------------- fragment loaders ----------------
__device__ __forceinline__ void load8_as_bf16(const __bf16* p, __bf16* d) {
    v8bf v = *(const v8bf*)p;                 // global_load_b128
#pragma unroll
    for (int i = 0; i < 8; ++i) d[i] = v[i];
}
__device__ __forceinline__ void load8_as_bf16(const float* p, __bf16* d) {
    const float4* q = (const float4*)p;
    float4 a = q[0], b = q[1];
    d[0] = (__bf16)a.x; d[1] = (__bf16)a.y; d[2] = (__bf16)a.z; d[3] = (__bf16)a.w;
    d[4] = (__bf16)b.x; d[5] = (__bf16)b.y; d[6] = (__bf16)b.z; d[7] = (__bf16)b.w;
}

// ---------------- WMMA GEMM: out = relu(A @ W + bias) ----------------
// A: [M x K] (fp32 or bf16), WT: [256 x K] bf16 (pre-transposed), out width = 256.
// Block = 128 threads = 4 waves; wave w covers N in [64w, 64w+64), block covers 16 rows.
// A-fragment layout per CDNA5 ISA 16-bit A 16x32: lane<16 -> K {0..7,16..23}, lane>=16 -> K {8..15,24..31}.
// B-fragment: lane<16 -> K {0..7,16..23}, lane>=16 -> K {8..15,24..31} (column = lane%16), from WT.
template <bool OUT_BF16, typename TA>
__global__ void gin_gemm_wmma(const TA* __restrict__ A,
                              const __bf16* __restrict__ WT,
                              const float* __restrict__ bias,
                              int K,
                              __bf16* __restrict__ out_bf,
                              float* __restrict__ out_f) {
    const int lane  = threadIdx.x & 31;
    const int wave  = threadIdx.x >> 5;
    const int l16   = lane & 15;
    const int hi    = lane >> 4;            // 0 or 1
    const int row0  = blockIdx.x * 16;
    const int nbase = wave * 64;

    const int arow  = row0 + l16;
    const int akoff = hi * 8;
    const int bkoff = hi * 8;

    v8f acc[4] = {};

    for (int kk = 0; kk < K; kk += 32) {
        // A fragment (convert fp32 -> bf16 in-register when TA == float)
        v16bf a;
        {
            __bf16 t0[8], t1[8];
            const TA* ap = A + (size_t)arow * K + kk + akoff;
            load8_as_bf16(ap, t0);
            load8_as_bf16(ap + 16, t1);
#pragma unroll
            for (int i = 0; i < 8; ++i) { a[i] = t0[i]; a[8 + i] = t1[i]; }
            if (kk + 32 < K)
                __builtin_prefetch(A + (size_t)arow * K + kk + 32 + akoff, 0, 3); // global_prefetch
        }
#pragma unroll
        for (int s = 0; s < 4; ++s) {
            const int col = nbase + s * 16 + l16;
            const __bf16* bp = WT + (size_t)col * K + kk + bkoff;
            v8bf b0 = *(const v8bf*)bp;
            v8bf b1 = *(const v8bf*)(bp + 16);
            v16bf b;
#pragma unroll
            for (int i = 0; i < 8; ++i) { b[i] = b0[i]; b[8 + i] = b1[i]; }
            acc[s] = __builtin_amdgcn_wmma_f32_16x16x32_bf16(
                false, a, false, b, (short)0, acc[s], false, false);
        }
    }

    // epilogue: bias + relu, store (bf16 for mid, fp32 for m)
#pragma unroll
    for (int s = 0; s < 4; ++s) {
        const int col = nbase + s * 16 + l16;
        const float bv = bias[col];
#pragma unroll
        for (int r = 0; r < 8; ++r) {
            const int row = row0 + r + hi * 8;
            float v = acc[s][r] + bv;
            v = v > 0.0f ? v : 0.0f;
            if (OUT_BF16) out_bf[(size_t)row * DIM + col] = (__bf16)v;
            else          out_f[(size_t)row * DIM + col] = v;
        }
    }
}

// ---------------- BatchNorm statistics ----------------
__global__ void bn_partial(const float* __restrict__ m,
                           float* __restrict__ sum, float* __restrict__ sumsq) {
    const int c = threadIdx.x;              // 256 columns, coalesced row reads
    float s = 0.0f, s2 = 0.0f;
    for (int r = blockIdx.x; r < N_NODES; r += gridDim.x) {
        float v = m[(size_t)r * DIM + c];
        s += v; s2 += v * v;
    }
    atomicAdd(&sum[c], s);
    atomicAdd(&sumsq[c], s2);
}

__global__ void bn_finalize(const float* __restrict__ sum, const float* __restrict__ sumsq,
                            const float* __restrict__ g, const float* __restrict__ b,
                            float* __restrict__ scale, float* __restrict__ shift) {
    const int c = threadIdx.x;
    const float inv_n = 1.0f / (float)N_NODES;
    float mu  = sum[c] * inv_n;
    float var = sumsq[c] * inv_n - mu * mu;   // biased variance (matches reference)
    float sc  = g[c] * rsqrtf(var + 1e-5f);
    scale[c] = sc;
    shift[c] = b[c] - mu * sc;
}

// ---------------- BN apply + concat-write + pooled readout ----------------
__global__ void bn_apply_pool(const float* __restrict__ m,
                              const float* __restrict__ scale, const float* __restrict__ shift,
                              const int* __restrict__ batch,
                              float* __restrict__ h_next,
                              float* __restrict__ out_xs,     // d_out xs region + layer*DIM
                              float* __restrict__ out_pool) { // d_out pool region + layer*DIM
    const int node = blockIdx.x;
    const int f    = threadIdx.x;
    float v = m[(size_t)node * DIM + f] * scale[f] + shift[f];
    h_next[(size_t)node * DIM + f] = v;
    out_xs[(size_t)node * OUT_DIM + f] = v;
    atomicAdd(&out_pool[(size_t)batch[node] * OUT_DIM + f], v);
}

// ---------------- host orchestration ----------------
extern "C" void kernel_launch(void* const* d_in, const int* in_sizes, int n_in,
                              void* d_out, int out_size, void* d_ws, size_t ws_size,
                              hipStream_t stream) {
    const float* x   = (const float*)d_in[0];
    const int*   ei  = (const int*)d_in[1];
    const int*   bat = (const int*)d_in[2];
    const int*   src = ei;
    const int*   dst = ei + N_EDGES;

    const float* w1[N_LAYERS]; const float* b1[N_LAYERS];
    const float* w2[N_LAYERS]; const float* b2[N_LAYERS];
    const float* bng[N_LAYERS]; const float* bnb[N_LAYERS];
    for (int i = 0; i < N_LAYERS; ++i) {
        w1[i]  = (const float*)d_in[3 + 6 * i + 0];
        b1[i]  = (const float*)d_in[3 + 6 * i + 1];
        w2[i]  = (const float*)d_in[3 + 6 * i + 2];
        b2[i]  = (const float*)d_in[3 + 6 * i + 3];
        bng[i] = (const float*)d_in[3 + 6 * i + 4];
        bnb[i] = (const float*)d_in[3 + 6 * i + 5];
    }

    // ---- workspace carve-up (all 256B-aligned sizes) ----
    char* ws = (char*)d_ws;
    size_t off = 0;
    float*  h_buf = (float*)(ws + off);  off += (size_t)N_NODES * DIM * 4;   // 51.2 MB
    float*  agg   = (float*)(ws + off);  off += (size_t)N_NODES * DIM * 4;   // 51.2 MB (reused for m)
    __bf16* mid   = (__bf16*)(ws + off); off += (size_t)N_NODES * DIM * 2;   // 25.6 MB
    __bf16* w1T[N_LAYERS]; __bf16* w2T[N_LAYERS];
    for (int i = 0; i < N_LAYERS; ++i) {
        int k1 = (i == 0) ? F_IN : DIM;
        w1T[i] = (__bf16*)(ws + off); off += (size_t)DIM * k1 * 2;
        w2T[i] = (__bf16*)(ws + off); off += (size_t)DIM * DIM * 2;
    }
    float* sum   = (float*)(ws + off); off += DIM * 4;
    float* sumsq = (float*)(ws + off); off += DIM * 4;
    float* scale = (float*)(ws + off); off += DIM * 4;
    float* shift = (float*)(ws + off); off += DIM * 4;

    float* out_pool = (float*)d_out;                               // [512 x 768]
    float* out_xs   = (float*)d_out + (size_t)N_GRAPHS * OUT_DIM;  // [50000 x 768]

    // zero pooled region (atomically accumulated -> must be zero every call)
    {
        long long n = (long long)N_GRAPHS * OUT_DIM;
        zero_f32<<<dim3((unsigned)((n + 255) / 256)), dim3(256), 0, stream>>>(out_pool, n);
    }

    // transpose+convert weights to bf16 [N x K]
    for (int i = 0; i < N_LAYERS; ++i) {
        int k1 = (i == 0) ? F_IN : DIM;
        wt_transpose_bf16<<<dim3((k1 * DIM + 255) / 256), dim3(256), 0, stream>>>(w1[i], w1T[i], k1);
        wt_transpose_bf16<<<dim3((DIM * DIM + 255) / 256), dim3(256), 0, stream>>>(w2[i], w2T[i], DIM);
    }

    const float* h = x;
    for (int i = 0; i < N_LAYERS; ++i) {
        const int fin  = (i == 0) ? F_IN : DIM;
        const int fin4 = fin / 4;

        // agg = h
        {
            int n4 = N_NODES * fin / 4;
            copy_f4<<<dim3((n4 + 255) / 256), dim3(256), 0, stream>>>(h, agg, n4);
        }
        // agg[dst] += h[src]
        {
            int total = N_EDGES * fin4;
            edge_scatter<<<dim3((total + 255) / 256), dim3(256), 0, stream>>>(
                h, src, dst, agg, fin4, total);
        }
        // mid = relu(agg @ w1 + b1)  (fp32 A -> bf16 WMMA, bf16 out)
        gin_gemm_wmma<true, float><<<dim3(N_NODES / 16), dim3(128), 0, stream>>>(
            agg, w1T[i], b1[i], fin, mid, nullptr);
        // m = relu(mid @ w2 + b2)    (bf16 A, fp32 out; reuse agg buffer as m)
        float* m = agg;
        gin_gemm_wmma<false, __bf16><<<dim3(N_NODES / 16), dim3(128), 0, stream>>>(
            mid, w2T[i], b2[i], DIM, nullptr, m);

        // BatchNorm statistics
        zero_f32<<<dim3(2), dim3(256), 0, stream>>>(sum, (long long)2 * DIM); // sum & sumsq contiguous
        bn_partial<<<dim3(512), dim3(DIM), 0, stream>>>(m, sum, sumsq);
        bn_finalize<<<dim3(1), dim3(DIM), 0, stream>>>(sum, sumsq, bng[i], bnb[i], scale, shift);

        // normalize + write concat output + pooled readout; h_buf becomes next h
        bn_apply_pool<<<dim3(N_NODES), dim3(DIM), 0, stream>>>(
            m, scale, shift, bat, h_buf, out_xs + (size_t)i * DIM, out_pool + (size_t)i * DIM);

        h = h_buf;
    }
}
